// TrajectoryGenerator_59579786330480
// MI455X (gfx1250) — compile-verified, matrix operand order checked by reading, not run on hardware
//
#include <hip/hip_runtime.h>

typedef __attribute__((ext_vector_type(16))) _Float16 v16h;
typedef __attribute__((ext_vector_type(8)))  float    v8f;

#define OBS_LEN 20
#define PRED_LEN 30
#define H 32
#define EMBD 16
#define MLPD 64
#define NOISE 8
#define G 32

// K index for VGPR pair j (each VGPR holds halves k,k+1), lane group grp (lane>>4).
// Per CDNA5 ISA 16-bit A layout: V0..V3 -> K=0..7 (+8 for lanes 16-31), V4..V7 -> K=16..23 (+8).
__device__ __forceinline__ int kdwm(int j, int grp) {
  int base = (j < 4) ? (2 * j) : (16 + 2 * (j - 4));
  return base + 8 * grp;
}

// Gather a 16x32 f16 A-tile row (this lane's row) from LDS into WMMA A register layout.
__device__ __forceinline__ v16h gatherA(const _Float16* row, int grp, int kofs) {
  v16h a;
#pragma unroll
  for (int j = 0; j < 8; ++j) {
    int k = kofs + kdwm(j, grp);
    a[2 * j]     = row[k];
    a[2 * j + 1] = row[k + 1];
  }
  return a;
}

// Load B-slice (K=32 x N=16) for out-column n from row-major f32 weight W[n][k] (ld = K stride).
// Rows n >= nmax are zero-padded (used for mlp2 24->32 pad).
__device__ __forceinline__ v16h loadB(const float* W, int n, int ld, int grp, int kofs, int nmax) {
  v16h b;
#pragma unroll
  for (int j = 0; j < 8; ++j) {
    int k = kofs + kdwm(j, grp);
    float w0 = 0.f, w1 = 0.f;
    if (n < nmax) { w0 = W[n * ld + k]; w1 = W[n * ld + k + 1]; }
    b[2 * j]     = (_Float16)w0;
    b[2 * j + 1] = (_Float16)w1;
  }
  return b;
}

__device__ __forceinline__ v8f wmma16(v16h a, v16h b, v8f c) {
  return __builtin_amdgcn_wmma_f32_16x16x32_f16(false, a, false, b, (short)0, c, false, false);
}

__device__ __forceinline__ float sigf(float x)  { return 1.f / (1.f + __expf(-x)); }
__device__ __forceinline__ float lrelu(float x) { return x > 0.f ? x : 0.01f * x; }

__global__ __launch_bounds__(256)
void sgan_scene_kernel(
    const float* __restrict__ obs_rel,                              // (20,N,2)
    const int*   __restrict__ se,                                   // (B,2)
    const float* __restrict__ noise_z,                              // (8)
    const float* __restrict__ enc_emb_w, const float* __restrict__ enc_emb_b,
    const float* __restrict__ enc_wih,   const float* __restrict__ enc_whh,
    const float* __restrict__ enc_bih,   const float* __restrict__ enc_bhh,
    const float* __restrict__ lne_g,     const float* __restrict__ lne_b,
    const float* __restrict__ wq, const float* __restrict__ wk,
    const float* __restrict__ wv, const float* __restrict__ wo,
    const float* __restrict__ lnc_g, const float* __restrict__ lnc_b,
    const float* __restrict__ mlp1_w, const float* __restrict__ mlp1_b,
    const float* __restrict__ mlp2_w, const float* __restrict__ mlp2_b,
    const float* __restrict__ dec_emb_w, const float* __restrict__ dec_emb_b,
    const float* __restrict__ dec_wih,   const float* __restrict__ dec_whh,
    const float* __restrict__ dec_bih,   const float* __restrict__ dec_bhh,
    const float* __restrict__ dec_out_w, const float* __restrict__ dec_out_b,
    float* __restrict__ preds, long long Nll)
{
  const long long N = Nll;

  __shared__ _Float16 h16[G * H];          // hidden state (enc h -> LN h -> dec h), f16 = WMMA A
  __shared__ float    cbuf[G * H];         // cell state
  __shared__ float    gbuf[G * 4 * H];     // 32x128 gate buffer / attention f32 scratch
  __shared__ float    qkv[3 * G * H];      // q,k,v (f32)
  __shared__ _Float16 m16a[G * 2 * H];     // f16 A scratch: attn_pre (32x32) then ctx (32x64)
  __shared__ _Float16 m16b[G * 2 * H];     // f16 A scratch: mlp1 output (32x64)
  __shared__ float    wfe[2 * 4 * H];      // folded enc input weights (128x2)
  __shared__ float    bfe[4 * H];
  __shared__ float    wfd[2 * 4 * H];      // folded dec input weights (128x2)
  __shared__ float    bfd[4 * H];
  __shared__ float    relb[G * 2];         // decoder last displacement
  __shared__ float    obsxy[G * 2];        // staged obs for one timestep

  const int tid  = threadIdx.x;
  const int lane = tid & 31;
  const int wave = tid >> 5;               // 0..7
  const int grp  = lane >> 4;              // 0/1
  const int ln   = lane & 15;              // column within 16-wide tile
  const int a0   = se[blockIdx.x * 2];     // first agent of this scene (G=32 agents)

  // zero h, c
  for (int i = tid; i < G * H; i += 256) { h16[i] = (_Float16)0.f; cbuf[i] = 0.f; }

  // Fold input path: gates_x = rel @ (Wih @ emb_w) + (Wih @ emb_b + bih + bhh)
  {
    int gi = tid & 127;
    const float* wih = (tid < 128) ? enc_wih   : dec_wih;
    const float* emw = (tid < 128) ? enc_emb_w : dec_emb_w;
    const float* emb = (tid < 128) ? enc_emb_b : dec_emb_b;
    const float* bih = (tid < 128) ? enc_bih   : dec_bih;
    const float* bhh = (tid < 128) ? enc_bhh   : dec_bhh;
    float w0 = 0.f, w1 = 0.f, bb = bih[gi] + bhh[gi];
#pragma unroll
    for (int e = 0; e < EMBD; ++e) {
      float wv_ = wih[gi * EMBD + e];
      w0 += wv_ * emw[e * 2 + 0];
      w1 += wv_ * emw[e * 2 + 1];
      bb += wv_ * emb[e];
    }
    if (tid < 128) { wfe[gi * 2] = w0; wfe[gi * 2 + 1] = w1; bfe[gi] = bb; }
    else           { wfd[gi * 2] = w0; wfd[gi * 2 + 1] = w1; bfd[gi] = bb; }
  }
  __syncthreads();

  const int ncol = wave * 16 + ln;         // this lane's gate column (0..127) in LSTM phases

  // ================= Encoder LSTM (20 steps) =================
  {
    v16h bEnc = loadB(enc_whh, ncol, H, grp, 0, 4 * H);   // Whh slice in regs for whole rollout
    const float wf0 = wfe[ncol * 2], wf1 = wfe[ncol * 2 + 1], bf0 = bfe[ncol];

    for (int t = 0; t < OBS_LEN; ++t) {
      if (tid < 64) obsxy[tid] = obs_rel[((size_t)t * N + a0) * 2 + tid];
      __syncthreads();
#pragma unroll
      for (int mt = 0; mt < 2; ++mt) {
        v8f c;
#pragma unroll
        for (int r = 0; r < 8; ++r) {
          int ag = mt * 16 + r + 8 * grp;
          c[r] = obsxy[ag * 2] * wf0 + obsxy[ag * 2 + 1] * wf1 + bf0;
        }
        v16h a = gatherA(&h16[(mt * 16 + ln) * H], grp, 0);
        v8f d  = wmma16(a, bEnc, c);
#pragma unroll
        for (int r = 0; r < 8; ++r)
          gbuf[(mt * 16 + r + 8 * grp) * (4 * H) + ncol] = d[r];
      }
      __syncthreads();
#pragma unroll
      for (int p = 0; p < 4; ++p) {
        int id = tid * 4 + p;
        int ag = id >> 5, u = id & 31;
        float gi_ = gbuf[ag * 128 + u];
        float gf_ = gbuf[ag * 128 + 32 + u];
        float gg_ = gbuf[ag * 128 + 64 + u];
        float go_ = gbuf[ag * 128 + 96 + u];
        float cc  = sigf(gf_) * cbuf[id] + sigf(gi_) * tanhf(gg_);
        float hh  = sigf(go_) * tanhf(cc);
        cbuf[id] = cc;
        h16[id]  = (_Float16)hh;
      }
      __syncthreads();
    }
  }

  // ================= LayerNorm(h_enc) =================
  if (tid < G) {
    float hv[H];
    float m = 0.f;
#pragma unroll
    for (int k = 0; k < H; ++k) { hv[k] = (float)h16[tid * H + k]; m += hv[k]; }
    m *= (1.f / H);
    float v = 0.f;
#pragma unroll
    for (int k = 0; k < H; ++k) { float d = hv[k] - m; v += d * d; }
    v *= (1.f / H);
    float rs = rsqrtf(v + 1e-5f);
#pragma unroll
    for (int k = 0; k < H; ++k)
      h16[tid * H + k] = (_Float16)((hv[k] - m) * rs * lne_g[k] + lne_b[k]);
  }
  __syncthreads();

  // ================= Attention: Q/K/V projections (WMMA) =================
  if (wave < 6) {
    int mat = wave >> 1, ntt = wave & 1;
    const float* Wp = (mat == 0) ? wq : (mat == 1) ? wk : wv;
    v16h bm = loadB(Wp, ntt * 16 + ln, H, grp, 0, H);
#pragma unroll
    for (int mt = 0; mt < 2; ++mt) {
      v16h a = gatherA(&h16[(mt * 16 + ln) * H], grp, 0);
      v8f c  = {};
      v8f d  = wmma16(a, bm, c);
#pragma unroll
      for (int r = 0; r < 8; ++r)
        qkv[mat * (G * H) + (mt * 16 + r + 8 * grp) * H + ntt * 16 + ln] = d[r];
    }
  }
  __syncthreads();

  // ---- scores + softmax + attn@V : 1 thread per (head, query), all registers ----
  if (tid < 128) {
    int hd = tid >> 5, qi = tid & 31;
    float qv_[8];
#pragma unroll
    for (int j = 0; j < 8; ++j) qv_[j] = qkv[qi * H + hd * 8 + j];
    float sc[G];
    float mx = -1e30f;
#pragma unroll
    for (int ky = 0; ky < G; ++ky) {
      float s = 0.f;
#pragma unroll
      for (int j = 0; j < 8; ++j) s += qv_[j] * qkv[G * H + ky * H + hd * 8 + j];
      s *= 0.35355339059f;   // 1/sqrt(8)
      sc[ky] = s;
      mx = fmaxf(mx, s);
    }
    float sum = 0.f;
    float ov[8] = {0, 0, 0, 0, 0, 0, 0, 0};
#pragma unroll
    for (int ky = 0; ky < G; ++ky) {
      float w = __expf(sc[ky] - mx);
      sum += w;
#pragma unroll
      for (int j = 0; j < 8; ++j) ov[j] += w * qkv[2 * G * H + ky * H + hd * 8 + j];
    }
    float inv = 1.f / sum;
#pragma unroll
    for (int j = 0; j < 8; ++j) m16a[qi * H + hd * 8 + j] = (_Float16)(ov[j] * inv);
  }
  __syncthreads();

  // ---- output projection attn @ wo.T (WMMA) ----
  if (wave < 4) {
    int mt = wave >> 1, ntt = wave & 1;
    v16h bm = loadB(wo, ntt * 16 + ln, H, grp, 0, H);
    v16h a  = gatherA(&m16a[(mt * 16 + ln) * H], grp, 0);
    v8f c   = {};
    v8f d   = wmma16(a, bm, c);
#pragma unroll
    for (int r = 0; r < 8; ++r)
      gbuf[G * H + (mt * 16 + r + 8 * grp) * H + ntt * 16 + ln] = d[r];
  }
  __syncthreads();

  // ================= ctx = LayerNorm(concat(h, attn)) -> f16 =================
  if (tid < G) {
    float cv[2 * H];
    float m = 0.f;
#pragma unroll
    for (int k = 0; k < H; ++k) { cv[k] = (float)h16[tid * H + k]; m += cv[k]; }
#pragma unroll
    for (int k = 0; k < H; ++k) { cv[H + k] = gbuf[G * H + tid * H + k]; m += cv[H + k]; }
    m *= (1.f / (2 * H));
    float v = 0.f;
#pragma unroll
    for (int k = 0; k < 2 * H; ++k) { float d = cv[k] - m; v += d * d; }
    v *= (1.f / (2 * H));
    float rs = rsqrtf(v + 1e-5f);
#pragma unroll
    for (int k = 0; k < 2 * H; ++k)
      m16a[tid * (2 * H) + k] = (_Float16)((cv[k] - m) * rs * lnc_g[k] + lnc_b[k]);
  }
  __syncthreads();

  // ================= MLP1: (32x64) @ (64x64) + LeakyReLU (WMMA, K split 2x32) =================
  {
    int mt = wave >> 2, ntt = wave & 3;
    int n  = ntt * 16 + ln;
    v16h b0 = loadB(mlp1_w, n, 2 * H, grp, 0, MLPD);
    v16h b1 = loadB(mlp1_w, n, 2 * H, grp, 32, MLPD);
    v8f c;
    float bv = mlp1_b[n];
#pragma unroll
    for (int r = 0; r < 8; ++r) c[r] = bv;
    v16h a0 = gatherA(&m16a[(mt * 16 + ln) * (2 * H)], grp, 0);
    v16h a1 = gatherA(&m16a[(mt * 16 + ln) * (2 * H)], grp, 32);
    v8f d = wmma16(a0, b0, c);
    d     = wmma16(a1, b1, d);
#pragma unroll
    for (int r = 0; r < 8; ++r)
      m16b[(mt * 16 + r + 8 * grp) * MLPD + n] = (_Float16)lrelu(d[r]);
  }
  __syncthreads();

  // ================= MLP2 (24 cols, padded to 32) + noise -> dec_h =================
  if (wave < 4) {
    int mt = wave >> 1, ntt = wave & 1;
    int n  = ntt * 16 + ln;
    v16h b0 = loadB(mlp2_w, n, 2 * H, grp, 0, H - NOISE);
    v16h b1 = loadB(mlp2_w, n, 2 * H, grp, 32, H - NOISE);
    v8f c;
    float bv = (n < H - NOISE) ? mlp2_b[n] : 0.f;
#pragma unroll
    for (int r = 0; r < 8; ++r) c[r] = bv;
    v16h a0 = gatherA(&m16b[(mt * 16 + ln) * MLPD], grp, 0);
    v16h a1 = gatherA(&m16b[(mt * 16 + ln) * MLPD], grp, 32);
    v8f d = wmma16(a0, b0, c);
    d     = wmma16(a1, b1, d);
#pragma unroll
    for (int r = 0; r < 8; ++r) {
      float val = (n < H - NOISE) ? lrelu(d[r]) : noise_z[n - (H - NOISE)];
      h16[(mt * 16 + r + 8 * grp) * H + n] = (_Float16)val;
    }
  }
  // reset cell state; seed last displacement with final observed step
  for (int i = tid; i < G * H; i += 256) cbuf[i] = 0.f;
  if (tid < 64) relb[tid] = obs_rel[((size_t)(OBS_LEN - 1) * N + a0) * 2 + tid];
  __syncthreads();

  // ================= Decoder LSTM rollout (30 steps) =================
  {
    v16h bDec = loadB(dec_whh, ncol, H, grp, 0, 4 * H);
    const float wd0 = wfd[ncol * 2], wd1 = wfd[ncol * 2 + 1], bd0 = bfd[ncol];

    for (int t = 0; t < PRED_LEN; ++t) {
#pragma unroll
      for (int mt = 0; mt < 2; ++mt) {
        v8f c;
#pragma unroll
        for (int r = 0; r < 8; ++r) {
          int ag = mt * 16 + r + 8 * grp;
          c[r] = relb[ag * 2] * wd0 + relb[ag * 2 + 1] * wd1 + bd0;
        }
        v16h a = gatherA(&h16[(mt * 16 + ln) * H], grp, 0);
        v8f d  = wmma16(a, bDec, c);
#pragma unroll
        for (int r = 0; r < 8; ++r)
          gbuf[(mt * 16 + r + 8 * grp) * (4 * H) + ncol] = d[r];
      }
      __syncthreads();
#pragma unroll
      for (int p = 0; p < 4; ++p) {
        int id = tid * 4 + p;
        int ag = id >> 5, u = id & 31;
        float gi_ = gbuf[ag * 128 + u];
        float gf_ = gbuf[ag * 128 + 32 + u];
        float gg_ = gbuf[ag * 128 + 64 + u];
        float go_ = gbuf[ag * 128 + 96 + u];
        float cc  = sigf(gf_) * cbuf[id] + sigf(gi_) * tanhf(gg_);
        float hh  = sigf(go_) * tanhf(cc);
        cbuf[id] = cc;
        h16[id]  = (_Float16)hh;
      }
      __syncthreads();
      if (tid < G) {
        float rx = dec_out_b[0], ry = dec_out_b[1];
#pragma unroll
        for (int k = 0; k < H; ++k) {
          float hv = (float)h16[tid * H + k];
          rx += hv * dec_out_w[k];
          ry += hv * dec_out_w[H + k];
        }
        relb[tid * 2]     = rx;
        relb[tid * 2 + 1] = ry;
        size_t ob = ((size_t)t * N + a0 + tid) * 2;
        preds[ob]     = rx;
        preds[ob + 1] = ry;
      }
      __syncthreads();
    }
  }
}

extern "C" void kernel_launch(void* const* d_in, const int* in_sizes, int n_in,
                              void* d_out, int out_size, void* d_ws, size_t ws_size,
                              hipStream_t stream) {
  (void)n_in; (void)out_size; (void)d_ws; (void)ws_size;
  const float* obs_rel   = (const float*)d_in[1];
  const int*   se        = (const int*)d_in[2];
  const float* noise_z   = (const float*)d_in[3];
  const float* enc_emb_w = (const float*)d_in[4];
  const float* enc_emb_b = (const float*)d_in[5];
  const float* enc_wih   = (const float*)d_in[6];
  const float* enc_whh   = (const float*)d_in[7];
  const float* enc_bih   = (const float*)d_in[8];
  const float* enc_bhh   = (const float*)d_in[9];
  const float* lne_g     = (const float*)d_in[10];
  const float* lne_b     = (const float*)d_in[11];
  const float* wq        = (const float*)d_in[12];
  const float* wk        = (const float*)d_in[13];
  const float* wv        = (const float*)d_in[14];
  const float* wo        = (const float*)d_in[15];
  const float* lnc_g     = (const float*)d_in[16];
  const float* lnc_b     = (const float*)d_in[17];
  const float* mlp1_w    = (const float*)d_in[18];
  const float* mlp1_b    = (const float*)d_in[19];
  const float* mlp2_w    = (const float*)d_in[20];
  const float* mlp2_b    = (const float*)d_in[21];
  const float* dec_emb_w = (const float*)d_in[22];
  const float* dec_emb_b = (const float*)d_in[23];
  const float* dec_wih   = (const float*)d_in[24];
  const float* dec_whh   = (const float*)d_in[25];
  const float* dec_bih   = (const float*)d_in[26];
  const float* dec_bhh   = (const float*)d_in[27];
  const float* dec_out_w = (const float*)d_in[28];
  const float* dec_out_b = (const float*)d_in[29];

  long long N      = (long long)in_sizes[1] / (OBS_LEN * 2);   // agents
  int numScenes    = in_sizes[2] / 2;                           // B (one block per scene of G=32)
  float* preds     = (float*)d_out;

  sgan_scene_kernel<<<numScenes, 256, 0, stream>>>(
      obs_rel, se, noise_z,
      enc_emb_w, enc_emb_b, enc_wih, enc_whh, enc_bih, enc_bhh,
      lne_g, lne_b, wq, wk, wv, wo, lnc_g, lnc_b,
      mlp1_w, mlp1_b, mlp2_w, mlp2_b,
      dec_emb_w, dec_emb_b, dec_wih, dec_whh, dec_bih, dec_bhh,
      dec_out_w, dec_out_b, preds, N);
}
